// MLA_30090540876004
// MI455X (gfx1250) — compile-verified, matrix operand order checked by reading, not run on hardware
//
#include <hip/hip_runtime.h>
#include <hip/hip_bf16.h>
#include <math.h>
#include <stdint.h>

// ---------------------------------------------------------------------------
// MLA forward for MI455X (gfx1250): wave32 + v_wmma_f32_16x16x32_bf16
// ---------------------------------------------------------------------------

typedef __bf16 bf16;
typedef __attribute__((ext_vector_type(16))) __bf16 v16bf;
typedef __attribute__((ext_vector_type(8)))  float  v8f;
typedef __attribute__((ext_vector_type(4)))  float  v4f;

#define B_   2
#define S_   2048
#define DIM_ 2048
#define H_   16
#define DN_  128
#define DR_  64
#define DV_  128
#define L_   512
#define M_   (B_ * S_)     // 4096 rows
#define QC_  576           // packed q/k width: 512 latent + 64 rope

__device__ __forceinline__ v8f wmma_bf16(v16bf a, v16bf b, v8f c) {
  return __builtin_amdgcn_wmma_f32_16x16x32_bf16(false, a, false, b, (short)0, c,
                                                 false, false);
}

union FragU { v16bf bf; v4f f4[2]; };

// Fragment load for 16x16x32 bf16 WMMA from a per-lane pointer p (16B aligned):
// elements 0..7 = p[0..7], elements 8..15 = p[16..23] (CDNA5 A/B layout with
// kbase = k0 + (lane>>4)*8 already folded into p by the caller).
__device__ __forceinline__ v16bf load_frag_p(const bf16* p) {
  FragU u;
  u.f4[0] = *(const v4f*)(p);
  u.f4[1] = *(const v4f*)(p + 16);
  return u.bf;
}

// ---------------------------------------------------------------------------
// Generic batched bf16 GEMM: C = A * B^T (+bias), one wave -> 64x64 C tile.
// A: (M,K) row-major, LDA. B: (N,K) row-major, LDB. Compile-time strides so
// row deltas fold into global_load immediate offsets (1 addr pair / matrix).
// grid = (N/64, M/64, batch), blockDim = 32. VGPRs capped at 256.
// ---------------------------------------------------------------------------
template <int LDA, int LDB, int LDC, bool F32OUT>
__global__ __launch_bounds__(32) __attribute__((amdgpu_num_vgpr(256)))
void gemm_wmma(const bf16* __restrict__ A, const bf16* __restrict__ Bm,
               float* __restrict__ Cf, bf16* __restrict__ Cb,
               const float* __restrict__ bias, int K,
               long sA, long sB, long sC) {
  A  += (size_t)blockIdx.z * (size_t)sA;
  Bm += (size_t)blockIdx.z * (size_t)sB;
  const int lane = threadIdx.x & 31;
  const int lrow = lane & 15;
  const int kh   = (lane >> 4) * 8;
  const int m0 = blockIdx.y * 64;
  const int n0 = blockIdx.x * 64;

  const bf16* Ap = A  + (size_t)(m0 + lrow) * LDA + kh;
  const bf16* Bp = Bm + (size_t)(n0 + lrow) * LDB + kh;

  v8f acc[4][4] = {};
  for (int k0 = 0; k0 < K; k0 += 32) {
    v16bf af[4], bfr[4];
#pragma unroll
    for (int mt = 0; mt < 4; mt++)
      af[mt] = load_frag_p(Ap + (size_t)mt * 16 * LDA + k0);
#pragma unroll
    for (int nt = 0; nt < 4; nt++)
      bfr[nt] = load_frag_p(Bp + (size_t)nt * 16 * LDB + k0);
#pragma unroll
    for (int mt = 0; mt < 4; mt++)
#pragma unroll
      for (int nt = 0; nt < 4; nt++)
        acc[mt][nt] = wmma_bf16(af[mt], bfr[nt], acc[mt][nt]);
  }

  const int rbase = (lane >> 4) * 8;
  const size_t crow = (size_t)blockIdx.z * (size_t)sC +
                      (size_t)(m0 + rbase) * LDC + (n0 + lrow);
#pragma unroll
  for (int nt = 0; nt < 4; nt++) {
    const float bv = bias ? bias[n0 + nt * 16 + lrow] : 0.f;
#pragma unroll
    for (int mt = 0; mt < 4; mt++)
#pragma unroll
      for (int i = 0; i < 8; i++) {
        const size_t off = crow + (size_t)(mt * 16 + i) * LDC + nt * 16;
        const float v = acc[mt][nt][i] + bv;
        if (F32OUT) Cf[off] = v;
        else        Cb[off] = (bf16)v;
      }
  }
}

// ---------------------------------------------------------------------------
// Elementwise kernels
// ---------------------------------------------------------------------------
__global__ void cvt_f32_bf16(const float* __restrict__ src, bf16* __restrict__ dst, long n) {
  long i = (long)blockIdx.x * 256 + threadIdx.x;
  if (i < n) dst[i] = (bf16)src[i];
}

// dst[h][c][d] = wkv_b_w[(h*256 + d)*512 + c]  (transposed nope half, bf16)
__global__ void wkvb_nope_T(const float* __restrict__ src, bf16* __restrict__ dst) {
  long i = (long)blockIdx.x * 256 + threadIdx.x;
  if (i >= (long)H_ * 512 * 128) return;
  int h   = (int)(i >> 16);
  int rem = (int)(i & 65535);
  int c = rem >> 7;
  int d = rem & 127;
  dst[i] = (bf16)src[((size_t)h * 256 + d) * 512 + c];
}

// Split q into q_nope (bf16, (M,H*128)) and roped q_pe -> qcat[...,512:576].
__global__ void q_post(const float* __restrict__ q, const float* __restrict__ fcos,
                       const float* __restrict__ fsin, bf16* __restrict__ qn,
                       bf16* __restrict__ qcat) {
  const int row = blockIdx.x;       // 0..M_-1  (= b*S + s)
  const int s   = row % S_;
  const int tid = threadIdx.x;      // blockDim = 128
  const float* qr = q + (size_t)row * (H_ * (DN_ + DR_));
  for (int idx = tid; idx < H_ * DN_; idx += 128) {
    int h = idx >> 7, d = idx & 127;
    qn[(size_t)row * (H_ * DN_) + h * DN_ + d] = (bf16)qr[h * 192 + d];
  }
  for (int idx = tid; idx < H_ * (DR_ / 2); idx += 128) {
    int h = idx >> 5, j = idx & 31;
    float re = qr[h * 192 + 128 + 2 * j];
    float im = qr[h * 192 + 128 + 2 * j + 1];
    float c  = fcos[s * 32 + j], sn = fsin[s * 32 + j];
    bf16* dst = qcat + ((size_t)row * H_ + h) * QC_ + 512;
    dst[2 * j]     = (bf16)(re * c - im * sn);
    dst[2 * j + 1] = (bf16)(re * sn + im * c);
  }
}

// LayerNorm latent (512) + rope k_pe (64) -> kcat row (576, bf16). 1 wave/row.
__global__ void kv_post(const float* __restrict__ kv, const float* __restrict__ g,
                        const float* __restrict__ bb, const float* __restrict__ fcos,
                        const float* __restrict__ fsin, bf16* __restrict__ kcat) {
  const int row = blockIdx.x;
  const int s   = row % S_;
  const int lane = threadIdx.x & 31;
  const float* kr = kv + (size_t)row * QC_;

  float sum = 0.f;
  for (int j = lane; j < 512; j += 32) sum += kr[j];
  sum += __shfl_xor(sum, 1, 32);  sum += __shfl_xor(sum, 2, 32);
  sum += __shfl_xor(sum, 4, 32);  sum += __shfl_xor(sum, 8, 32);
  sum += __shfl_xor(sum, 16, 32);
  const float mu = sum * (1.f / 512.f);

  float vs = 0.f;
  for (int j = lane; j < 512; j += 32) { float d = kr[j] - mu; vs += d * d; }
  vs += __shfl_xor(vs, 1, 32);  vs += __shfl_xor(vs, 2, 32);
  vs += __shfl_xor(vs, 4, 32);  vs += __shfl_xor(vs, 8, 32);
  vs += __shfl_xor(vs, 16, 32);
  const float rs = rsqrtf(vs * (1.f / 512.f) + 1e-5f);

  bf16* dst = kcat + (size_t)row * QC_;
  for (int j = lane; j < 512; j += 32)
    dst[j] = (bf16)((kr[j] - mu) * rs * g[j] + bb[j]);

  { // rope k_pe: 32 pairs, one per lane
    int j = lane;
    float re = kr[512 + 2 * j], im = kr[512 + 2 * j + 1];
    float c  = fcos[s * 32 + j], sn = fsin[s * 32 + j];
    dst[512 + 2 * j]     = (bf16)(re * c - im * sn);
    dst[512 + 2 * j + 1] = (bf16)(re * sn + im * c);
  }
}

// ---------------------------------------------------------------------------
// Fused flash attention. 2 waves / block, block handles 16 query rows for one
// (b, h). The 16x576 Q tile is staged into LDS once (so the loop-invariant
// A-fragments come from ds_load_b128, not registers/scratch). Wave w computes
// score columns [w*32, w*32+32) (K=576 WMMA) and owns ctx columns
// {ph*256 + w*128 .. +128}. Softmax stats merged across waves via LDS; P and
// V^T staged through LDS for the P@V WMMA.
// grid = (S/16, H, B), blockDim = 64. VGPRs capped at 256 (no MSB churn).
// ---------------------------------------------------------------------------
#define LDT 72   // padded t-stride for the V^T LDS slab (multiple of 8)

__global__ __launch_bounds__(64) __attribute__((amdgpu_num_vgpr(256)))
void mla_attn(const bf16* __restrict__ qcat, const bf16* __restrict__ kcat,
              bf16* __restrict__ ctx, const int* __restrict__ spp) {
  __shared__ __align__(16) bf16 qlds[16 * QC_];   // Q tile: 18,432 B
  __shared__ __align__(16) bf16 ldsT[256 * LDT];  // V^T slice: 36,864 B
  __shared__ __align__(16) bf16 plds[16 * 64];    // P tile: 2,048 B
  __shared__ float mpart[2][16];
  __shared__ float lpart[2][16];

  const int qt = blockIdx.x, h = blockIdx.y, b = blockIdx.z;
  const int qs = qt * 16;
  const int sp = spp[0];
  const int tid  = threadIdx.x;
  const int w    = tid >> 5;          // wave id: 0 or 1
  const int lane = tid & 31;
  const int lrow = lane & 15;
  const int kh   = (lane >> 4) * 8;
  const float scale = 0.07216878364870322f;  // (DN+DR)^-0.5
  const float NEG_INF = -__builtin_inff();

  // ---- stage Q tile into LDS: qlds[r][c] = qcat[b, qs+r, h, c] ----
  {
    const bf16* qsrc = qcat + ((size_t)(b * S_ + qs) * H_ + h) * QC_;
    for (int idx = tid; idx < 16 * (QC_ / 8); idx += 64) {   // 1152 v4f groups
      int r  = idx / (QC_ / 8);
      int cg = (idx % (QC_ / 8)) * 8;
      *(v4f*)(qlds + r * QC_ + cg) =
          *(const v4f*)(qsrc + (size_t)r * (H_ * QC_) + cg);
    }
  }
  __syncthreads();

  const bf16* kbase = kcat + (size_t)b * S_ * QC_;
  const bf16* qfrag = qlds + lrow * QC_ + kh;   // per-lane A-frag base in LDS

  v8f cacc[16] = {};                  // this wave's ctx: 16 rows x 256 cols
  float mrow[8], lsum[8];
#pragma unroll
  for (int i = 0; i < 8; i++) { mrow[i] = -3.0e38f; lsum[i] = 0.f; }

  int tmax = qs + 15 + sp;
  if (tmax > S_ - 1) tmax = S_ - 1;

  for (int t0 = 0; t0 <= tmax; t0 += 64) {
    // ---- scores: this wave's 16 x 32 slab, K = 576 ----
    v8f s0 = {}, s1 = {};
    const bf16* k0p = kbase + (size_t)(t0 + w * 32 + lrow) * QC_ + kh;
    const bf16* k1p = k0p + (size_t)16 * QC_;
    for (int ks = 0; ks < 18; ks++) {
      v16bf qa  = load_frag_p(qfrag + ks * 32);
      v16bf kf0 = load_frag_p(k0p + ks * 32);
      v16bf kf1 = load_frag_p(k1p + ks * 32);
      s0 = wmma_bf16(qa, kf0, s0);
      s1 = wmma_bf16(qa, kf1, s1);
    }
    // scale + causal mask (col for s0: t0+w*32+lrow, s1: +16)
#pragma unroll
    for (int i = 0; i < 8; i++) {
      const int rg = qs + kh + i + sp;          // qpos
      const int c0 = t0 + w * 32 + lrow;        // tpos
      float v0 = s0[i] * scale, v1 = s1[i] * scale;
      s0[i] = (rg >= c0)      ? v0 : NEG_INF;
      s1[i] = (rg >= c0 + 16) ? v1 : NEG_INF;
    }
    // ---- cross-wave online softmax ----
    float mt8[8];
#pragma unroll
    for (int i = 0; i < 8; i++) {
      float v = fmaxf(s0[i], s1[i]);
      v = fmaxf(v, __shfl_xor(v, 1, 32));
      v = fmaxf(v, __shfl_xor(v, 2, 32));
      v = fmaxf(v, __shfl_xor(v, 4, 32));
      v = fmaxf(v, __shfl_xor(v, 8, 32));
      mt8[i] = v;
    }
    if (lrow == 0) {
#pragma unroll
      for (int i = 0; i < 8; i++) mpart[w][kh + i] = mt8[i];
    }
    __syncthreads();
    float corr[8];
#pragma unroll
    for (int i = 0; i < 8; i++) {
      float mn = fmaxf(mrow[i], fmaxf(mpart[0][kh + i], mpart[1][kh + i]));
      corr[i] = __expf(mrow[i] - mn);
      mrow[i] = mn;
    }
#pragma unroll
    for (int c = 0; c < 16; c++)
#pragma unroll
      for (int i = 0; i < 8; i++) cacc[c][i] *= corr[i];

    float rs8[8];
#pragma unroll
    for (int i = 0; i < 8; i++) {
      float p0 = __expf(s0[i] - mrow[i]);
      float p1 = __expf(s1[i] - mrow[i]);
      plds[(kh + i) * 64 + w * 32 + lrow]      = (bf16)p0;
      plds[(kh + i) * 64 + w * 32 + 16 + lrow] = (bf16)p1;
      float v = p0 + p1;
      v += __shfl_xor(v, 1, 32);
      v += __shfl_xor(v, 2, 32);
      v += __shfl_xor(v, 4, 32);
      v += __shfl_xor(v, 8, 32);
      rs8[i] = v;
    }
    if (lrow == 0) {
#pragma unroll
      for (int i = 0; i < 8; i++) lpart[w][kh + i] = rs8[i];
    }
    __syncthreads();
#pragma unroll
    for (int i = 0; i < 8; i++)
      lsum[i] = lsum[i] * corr[i] + lpart[0][kh + i] + lpart[1][kh + i];

    // ---- ctx += P(16x64) @ V(64x512), two 256-col phases ----
#pragma unroll
    for (int ph = 0; ph < 2; ph++) {
      // stage V^T slice: ldsT[c][t] = kvn[t0+t][ph*256+c], 64 threads
      for (int idx = tid; idx < 64 * 32; idx += 64) {
        int t  = idx >> 5;
        int cg = (idx & 31) * 8;
        const bf16* src = kcat + ((size_t)b * S_ + t0 + t) * QC_ + ph * 256 + cg;
        FragU u;
        u.f4[0] = *(const v4f*)src;
#pragma unroll
        for (int e = 0; e < 8; e++) ldsT[(cg + e) * LDT + t] = u.bf[e];
      }
      __syncthreads();
#pragma unroll
      for (int kt = 0; kt < 2; kt++) {
        v16bf pf = load_frag_p(plds + lrow * 64 + kt * 32 + kh);
#pragma unroll
        for (int nc = 0; nc < 8; nc++) {
          v16bf vf = load_frag_p(ldsT + (size_t)(w * 128 + nc * 16 + lrow) * LDT +
                                 kt * 32 + kh);
          cacc[ph * 8 + nc] = wmma_bf16(pf, vf, cacc[ph * 8 + nc]);
        }
      }
      __syncthreads();
    }
  }

  // ---- normalize + store this wave's ctx cols (bf16, layout (b,s,h,512)) ----
#pragma unroll
  for (int c = 0; c < 16; c++) {
    const int col = ((c >> 3) * 256) + (w * 128) + ((c & 7) * 16) + lrow;
#pragma unroll
    for (int i = 0; i < 8; i++) {
      const int rg = qs + kh + i;
      const float v = cacc[c][i] / lsum[i];
      ctx[(((size_t)b * S_ + rg) * H_ + h) * L_ + col] = (bf16)v;
    }
  }
}

// ---------------------------------------------------------------------------
// Host-side orchestration
// ---------------------------------------------------------------------------
extern "C" void kernel_launch(void* const* d_in, const int* in_sizes, int n_in,
                              void* d_out, int out_size, void* d_ws, size_t ws_size,
                              hipStream_t stream) {
  (void)in_sizes; (void)n_in; (void)out_size; (void)ws_size;
  const float* x      = (const float*)d_in[0];
  const int*   sp     = (const int*)  d_in[1];
  const float* fcos   = (const float*)d_in[2];
  const float* fsin   = (const float*)d_in[3];
  const float* wq_w   = (const float*)d_in[4];
  const float* wq_b   = (const float*)d_in[5];
  const float* wkva_w = (const float*)d_in[6];
  const float* wkva_b = (const float*)d_in[7];
  const float* kng    = (const float*)d_in[8];
  const float* knb    = (const float*)d_in[9];
  const float* wkvb_w = (const float*)d_in[10];
  const float* wo_w   = (const float*)d_in[11];
  const float* wo_b   = (const float*)d_in[12];
  float* out = (float*)d_out;

  char* ws = (char*)d_ws;
  size_t off = 0;
  auto alloc = [&](size_t bytes) -> void* {
    void* p = ws + off;
    off = (off + bytes + 255) & ~(size_t)255;
    return p;
  };

  bf16*  xb     = (bf16*) alloc((size_t)M_ * DIM_ * 2);
  bf16*  wq16   = (bf16*) alloc((size_t)3072 * 2048 * 2);
  bf16*  wkva16 = (bf16*) alloc((size_t)576 * 2048 * 2);
  bf16*  wkvb16 = (bf16*) alloc((size_t)4096 * 512 * 2);
  bf16*  wkvbnT = (bf16*) alloc((size_t)H_ * 512 * 128 * 2);
  bf16*  wo16   = (bf16*) alloc((size_t)2048 * 2048 * 2);
  float* qf     = (float*)alloc((size_t)M_ * 3072 * 4);
  float* kvf    = (float*)alloc((size_t)M_ * 576 * 4);
  bf16*  qn     = (bf16*) alloc((size_t)M_ * 2048 * 2);
  bf16*  qcat   = (bf16*) alloc((size_t)M_ * H_ * QC_ * 2);
  bf16*  kcat   = (bf16*) alloc((size_t)M_ * QC_ * 2);
  bf16*  ctxb   = (bf16*) alloc((size_t)M_ * H_ * L_ * 2);
  bf16*  attn   = (bf16*) alloc((size_t)M_ * 2048 * 2);

  auto cvt = [&](const float* s, bf16* d, long n) {
    cvt_f32_bf16<<<dim3((unsigned)((n + 255) / 256)), dim3(256), 0, stream>>>(s, d, n);
  };
  cvt(x,      xb,     (long)M_ * DIM_);
  cvt(wq_w,   wq16,   3072L * 2048);
  cvt(wkva_w, wkva16, 576L * 2048);
  cvt(wkvb_w, wkvb16, 4096L * 512);
  cvt(wo_w,   wo16,   2048L * 2048);
  wkvb_nope_T<<<dim3((H_ * 512 * 128 + 255) / 256), 256, 0, stream>>>(wkvb_w, wkvbnT);

  // q = x @ wq^T + wq_b            (4096 x 3072, K=2048)
  gemm_wmma<2048, 2048, 3072, true><<<dim3(3072 / 64, M_ / 64, 1), 32, 0, stream>>>(
      xb, wq16, qf, nullptr, wq_b, 2048, 0, 0, 0);
  // kv_full = x @ wkva^T + b       (4096 x 576, K=2048)
  gemm_wmma<2048, 2048, 576, true><<<dim3(576 / 64, M_ / 64, 1), 32, 0, stream>>>(
      xb, wkva16, kvf, nullptr, wkva_b, 2048, 0, 0, 0);

  q_post<<<dim3(M_), 128, 0, stream>>>(qf, fcos, fsin, qn, qcat);
  kv_post<<<dim3(M_), 32, 0, stream>>>(kvf, kng, knb, fcos, fsin, kcat);

  // q_abs[s,h,:] = q_nope[s,h,:] @ wkv_b_nope[h]^T  -> qcat[...,0:512]
  // batched over h: A stride 128, B stride 512*128, C stride 576
  gemm_wmma<2048, 128, H_ * QC_, false><<<dim3(512 / 64, M_ / 64, H_), 32, 0, stream>>>(
      qn, wkvbnT, nullptr, qcat, nullptr, 128, 128, 512L * 128, QC_);

  // fused flash attention
  mla_attn<<<dim3(S_ / 16, H_, B_), 64, 0, stream>>>(qcat, kcat, ctxb, sp);

  // out_h = ctx[s,h,:] @ wkv_b_v[h]^T  (N=128, K=512), gathered to (M, H*128)
  gemm_wmma<H_ * L_, 512, 2048, false><<<dim3(128 / 64, M_ / 64, H_), 32, 0, stream>>>(
      ctxb, wkvb16 + 128L * 512, nullptr, attn, nullptr, 512, 512, 256L * 512, 128);

  // final: out = attn @ wo^T + wo_b   (4096 x 2048, K=2048), fp32 output
  gemm_wmma<2048, 2048, 2048, true><<<dim3(2048 / 64, M_ / 64, 1), 32, 0, stream>>>(
      attn, wo16, out, nullptr, wo_b, 2048, 0, 0, 0);
}
